// StructuredPredictor_70523363000725
// MI455X (gfx1250) — matrix operand order repaired and checked by reading, not compiled, not run on hardware
//
#include <hip/hip_runtime.h>
#include <hip/hip_bf16.h>

typedef __attribute__((ext_vector_type(16))) _Float16 v16h;
typedef __attribute__((ext_vector_type(8)))  _Float16 v8h;
typedef __attribute__((ext_vector_type(8)))  float    v8f;

// ---------------- pack / cast kernels ----------------

// x NCHW f32 (256,12,128,128) -> NHWC f16 padded to 16 channels
__global__ void cast_to_nhwc(const float* __restrict__ src, _Float16* __restrict__ dst) {
    int i = blockIdx.x * blockDim.x + threadIdx.x;    // x-fastest for coalesced reads
    int x = i & 127;
    int c = (i >> 7) & 15;
    int y = (i >> 11) & 127;
    int b = i >> 18;
    _Float16 v = (_Float16)0.0f;
    if (c < 12) v = (_Float16)src[(((size_t)b * 12 + c) * 128 + y) * 128 + x];
    dst[(((size_t)b * 128 + y) * 128 + x) * 16 + c] = v;
}

// conv weights OIHW -> [Cout][Kp] f16 with k = tap*Cpad + cin (tap = ky*3+kx), zeros for pad
__global__ void pack_conv_w(const float* __restrict__ src, _Float16* __restrict__ dst,
                            int Cout, int Cin, int Kp, int lc) {
    int i = blockIdx.x * blockDim.x + threadIdx.x;
    if (i >= Cout * Kp) return;
    int co  = i / Kp;
    int k   = i - co * Kp;
    int tap = k >> lc;
    int cin = k & ((1 << lc) - 1);
    _Float16 v = (_Float16)0.0f;
    if (tap < 9 && cin < Cin)
        v = (_Float16)src[((size_t)co * Cin + cin) * 9 + tap];
    dst[i] = v;
}

// wl1 [128][4096] -> fragment-linear B panels, with NCHW-flatten -> NHWC-flatten permutation.
__global__ void pack_fc1_w(const float* __restrict__ src, _Float16* __restrict__ dst) {
    int i = blockIdx.x * blockDim.x + threadIdx.x;
    if (i >= 4096 * 128) return;
    int e    = i & 15;
    int lane = (i >> 4) & 31;
    int nt   = (i >> 9) & 7;
    int kc32 = i >> 12;
    int K    = kc32 * 32 + (lane >> 4) * 16 + e;  // our feature index (s*64 + c)
    int n    = nt * 16 + (lane & 15);
    int c    = K & 63;
    int s    = K >> 6;
    dst[i] = (_Float16)src[(size_t)n * 4096 + c * 64 + s];
}

// ---------------- fused conv3x3(SAME)+bias+relu+maxpool2, NHWC, WMMA ----------------
// grid: (ceil(W/32), H/2, B)  block: (Cout/16)*2 waves
// K ordering: tap*CPAD + cin ; LC compile-time -> full unroll, immediate-offset DS loads
template<int LC>
__global__ void conv_pool_wmma(const _Float16* __restrict__ in, const _Float16* __restrict__ wp,
                               const float* __restrict__ bias, _Float16* __restrict__ out,
                               int Cout, int H, int W) {
    constexpr int CPAD  = 1 << LC;
    constexpr int CMASK = CPAD - 1;
    constexpr int RSTR  = 34 << LC;                 // LDS row stride (halves)
    constexpr int KP    = (LC == 4) ? 160 : (9 << LC);
    constexpr uint64_t TAB = 0x4544242322020100ULL; // tap -> ky*34+kx, taps 0..7 (tap8 = 70)

    __shared__ _Float16 smem[4 * 34 * 64];          // [r][x][cin]

    const int lane = threadIdx.x & 31;
    const int wave = threadIdx.x >> 5;
    const int chT  = wave >> 1;
    const int sw   = wave & 1;
    const int b    = blockIdx.z;
    const int hp   = blockIdx.y;
    const int h0   = hp * 2;
    const int w0   = blockIdx.x * 32;

    // ---- stage NHWC strip into LDS as v8h copies (coalesced) ----
    constexpr int CSEG = CPAD >> 3;
    constexpr int NVEC = 4 * 34 * CSEG;
    for (int i = threadIdx.x; i < NVEC; i += blockDim.x) {
        int cs  = i & (CSEG - 1);
        int pix = i >> (LC - 3);
        int x   = pix % 34;
        int r   = pix / 34;
        int gr  = h0 - 1 + r;
        int gc  = w0 - 1 + x;
        v8h v = {};
        if (gr >= 0 && gr < H && gc >= 0 && gc < W)
            v = *(const v8h*)(in + (((size_t)b * H + gr) * W + gc) * CPAD + (cs << 3));
        *(v8h*)(smem + ((r * 34 + x) << LC) + (cs << 3)) = v;
    }
    __syncthreads();

    const int l15      = lane & 15;
    const int hi       = lane >> 4;
    const int m0       = chT * 16;
    const int jlocal   = sw * 16 + l15;
    const int laneBase = jlocal << LC;

    v8f acc0 = {};   // conv row h0
    v8f acc1 = {};   // conv row h0+1
    const _Float16* wbase = wp + (size_t)(m0 + l15) * KP;

#pragma unroll
    for (int kc = 0; kc < KP; kc += 32) {
        // A fragment: two contiguous b128 loads (immediate offsets from wbase)
        v8h a_lo = *(const v8h*)(wbase + kc + hi * 16);
        v8h a_hi = *(const v8h*)(wbase + kc + hi * 16 + 8);
        v16h afrag;
#pragma unroll
        for (int e = 0; e < 8; ++e) { afrag[e] = a_lo[e]; afrag[8 + e] = a_hi[e]; }

        // B addressing: wave-uniform per half-chunk, constant-folded after unroll
        int tapL = kc >> LC;
        int tapH = (kc + 16) >> LC;
        int offL = (tapL >= 8) ? 70 : (int)((TAB >> ((tapL & 7) * 8)) & 0xff);
        int offH = (tapH >= 8) ? 70 : (int)((TAB >> ((tapH & 7) * 8)) & 0xff);
        offL = (offL << LC) + (kc & CMASK);
        offH = (offH << LC) + ((kc + 16) & CMASK);
        int a0 = laneBase + (hi ? offH : offL);

        v8h p0 = *(const v8h*)(smem + a0);
        v8h p1 = *(const v8h*)(smem + a0 + 8);
        v8h q0 = *(const v8h*)(smem + a0 + RSTR);
        v8h q1 = *(const v8h*)(smem + a0 + RSTR + 8);
        v16h b0, b1;
#pragma unroll
        for (int e = 0; e < 8; ++e) {
            b0[e] = p0[e]; b0[8 + e] = p1[e];
            b1[e] = q0[e]; b1[8 + e] = q1[e];
        }

        acc0 = __builtin_amdgcn_wmma_f32_16x16x32_f16(false, afrag, false, b0, (short)0, acc0, false, false);
        acc1 = __builtin_amdgcn_wmma_f32_16x16x32_f16(false, afrag, false, b1, (short)0, acc1, false, false);
    }

    // ---- bias + 2x2 maxpool + relu, pack 8 channels, one b128 store per even lane ----
    const int Hout = H >> 1, Wout = W >> 1;
    v8h ov;
#pragma unroll
    for (int r = 0; r < 8; ++r) {
        float bs = bias[m0 + hi * 8 + r];
        float vv = fmaxf(acc0[r] + bs, acc1[r] + bs);     // vertical pool
        float o  = fmaxf(vv, __shfl_xor(vv, 1, 32));      // horizontal pool (lane pairs)
        ov[r] = (_Float16)fmaxf(o, 0.0f);
    }
    if ((lane & 1) == 0) {
        int pcol = (w0 + jlocal) >> 1;
        if (pcol < Wout)
            *(v8h*)(out + ((((size_t)b * Hout + hp) * Wout + pcol) * Cout) + m0 + hi * 8) = ov;
    }
}

// ---------------- FC1: (256 x 4096) @ packed W + bias, relu -> f32 ----------------
__global__ void fc1_wmma(const _Float16* __restrict__ act, const _Float16* __restrict__ wfrag,
                         const float* __restrict__ bias, float* __restrict__ hout) {
    const int lane = threadIdx.x & 31;
    const int nt   = threadIdx.x >> 5;
    const int mt   = blockIdx.x;
    const int l15  = lane & 15;
    const int hi   = lane >> 4;

    v8f acc = {};
    const _Float16* arow  = act + (size_t)(mt * 16 + l15) * 4096;
    const _Float16* bbase = wfrag + nt * 512 + lane * 16;   // + kc*128 per chunk (linear)
    const int ncol = nt * 16 + l15;

#pragma unroll 8
    for (int kc = 0; kc < 4096; kc += 32) {
        v8h a_lo = *(const v8h*)(arow + kc + hi * 16);
        v8h a_hi = *(const v8h*)(arow + kc + hi * 16 + 8);
        const _Float16* bp = bbase + (size_t)kc * 128;
        v8h b_lo = *(const v8h*)(bp);
        v8h b_hi = *(const v8h*)(bp + 8);
        v16h afrag, bfrag;
#pragma unroll
        for (int e = 0; e < 8; ++e) {
            afrag[e] = a_lo[e]; afrag[8 + e] = a_hi[e];
            bfrag[e] = b_lo[e]; bfrag[8 + e] = b_hi[e];
        }
        acc = __builtin_amdgcn_wmma_f32_16x16x32_f16(false, afrag, false, bfrag, (short)0, acc, false, false);
    }

#pragma unroll
    for (int r = 0; r < 8; ++r) {
        int row = mt * 16 + r + hi * 8;
        hout[(size_t)row * 128 + ncol] = fmaxf(acc[r] + bias[ncol], 0.0f);
    }
}

// ---------------- head: state = h@wl2.T + bl2 ; fx = state0 + 20*state2 ----------------
__global__ void head_kernel(const float* __restrict__ h, const float* __restrict__ wl2,
                            const float* __restrict__ bl2, float* __restrict__ out) {
    int b = blockIdx.x * blockDim.x + threadIdx.x;
    if (b >= 256) return;
    float a0 = 0.f, a2 = 0.f;
    const float* hr = h + (size_t)b * 128;
#pragma unroll 4
    for (int k = 0; k < 128; ++k) {
        float hv = hr[k];
        a0 += hv * wl2[k];
        a2 += hv * wl2[2 * 128 + k];
    }
    out[b] = (a0 + bl2[0]) + 20.0f * (a2 + bl2[2]);
}

// ---------------- launcher ----------------
extern "C" void kernel_launch(void* const* d_in, const int* in_sizes, int n_in,
                              void* d_out, int out_size, void* d_ws, size_t ws_size,
                              hipStream_t stream) {
    const float* x   = (const float*)d_in[0];
    const float* w1  = (const float*)d_in[1];  const float* b1  = (const float*)d_in[2];
    const float* w2  = (const float*)d_in[3];  const float* b2  = (const float*)d_in[4];
    const float* w3  = (const float*)d_in[5];  const float* b3  = (const float*)d_in[6];
    const float* w4  = (const float*)d_in[7];  const float* b4  = (const float*)d_in[8];
    const float* wl1 = (const float*)d_in[9];  const float* bl1 = (const float*)d_in[10];
    const float* wl2 = (const float*)d_in[11]; const float* bl2 = (const float*)d_in[12];
    float* out = (float*)d_out;

    char* ws = (char*)d_ws;
    size_t off = 0;
    auto alloc = [&](size_t bytes) { size_t o = off; off = (off + bytes + 255) & ~255ULL; return o; };

    _Float16* X0   = (_Float16*)(ws + alloc((size_t)256 * 128 * 128 * 16 * 2)); // NHWC pad16
    _Float16* PA   = (_Float16*)(ws + alloc((size_t)256 * 64 * 64 * 32 * 2));   // pool1 / pool3
    _Float16* PB   = (_Float16*)(ws + alloc((size_t)256 * 32 * 32 * 64 * 2));   // pool2 / pool4
    _Float16* Wp1  = (_Float16*)(ws + alloc((size_t)32 * 160 * 2));
    _Float16* Wp2  = (_Float16*)(ws + alloc((size_t)64 * 288 * 2));
    _Float16* Wp3  = (_Float16*)(ws + alloc((size_t)64 * 576 * 2));
    _Float16* Wp4  = (_Float16*)(ws + alloc((size_t)64 * 576 * 2));
    _Float16* WF   = (_Float16*)(ws + alloc((size_t)4096 * 128 * 2));
    float*    Hbuf = (float*)   (ws + alloc((size_t)256 * 128 * 4));

    {   // cast + pad input to NHWC f16
        int n = 256 * 128 * 128 * 16;
        cast_to_nhwc<<<n / 256, 256, 0, stream>>>(x, X0);
    }
    pack_conv_w<<<(32 * 160 + 255) / 256, 256, 0, stream>>>(w1, Wp1, 32, 12, 160, 4);
    pack_conv_w<<<(64 * 288 + 255) / 256, 256, 0, stream>>>(w2, Wp2, 64, 32, 288, 5);
    pack_conv_w<<<(64 * 576 + 255) / 256, 256, 0, stream>>>(w3, Wp3, 64, 64, 576, 6);
    pack_conv_w<<<(64 * 576 + 255) / 256, 256, 0, stream>>>(w4, Wp4, 64, 64, 576, 6);
    pack_fc1_w<<<(4096 * 128 + 255) / 256, 256, 0, stream>>>(wl1, WF);

    // conv layers (fused bias+relu+maxpool), NHWC f16
    conv_pool_wmma<4><<<dim3(4, 64, 256), 128, 0, stream>>>(X0, Wp1, b1, PA, 32, 128, 128);
    conv_pool_wmma<5><<<dim3(2, 32, 256), 256, 0, stream>>>(PA, Wp2, b2, PB, 64,  64,  64);
    conv_pool_wmma<6><<<dim3(1, 16, 256), 256, 0, stream>>>(PB, Wp3, b3, PA, 64,  32,  32);
    conv_pool_wmma<6><<<dim3(1,  8, 256), 256, 0, stream>>>(PA, Wp4, b4, PB, 64,  16,  16);

    // fc1 + relu (activation layout matches packed-W permutation)
    fc1_wmma<<<16, 256, 0, stream>>>(PB, WF, bl1, Hbuf);

    // fc2 + 200-step ballistic scan collapsed to px + 20*vx
    head_kernel<<<1, 256, 0, stream>>>(Hbuf, wl2, bl2, out);

    (void)in_sizes; (void)n_in; (void)out_size; (void)ws_size;
}
